// Encoder_64338610094319
// MI455X (gfx1250) — compile-verified, matrix operand order checked by reading, not run on hardware
//
#include <hip/hip_runtime.h>
#include <hip/hip_bf16.h>
#include <math.h>

// Problem constants (match reference)
#define TT 512
#define BB 32
#define EE 512
#define DD 512
#define PAD_IDX 3
#define M_ROWS (TT * BB)      // 16384
#define N_U (4 * DD)          // 2048
#define N_OUT DD              // 512
#define K_OUT (2 * DD)        // 1024

typedef __attribute__((ext_vector_type(16))) _Float16 v16h;
typedef __attribute__((ext_vector_type(8)))  _Float16 v8h;
typedef __attribute__((ext_vector_type(8)))  float    v8f;
typedef __attribute__((ext_vector_type(4)))  float    v4f;

// ---------------------------------------------------------------------------
// Fragment load for 16-bit A/B operands of v_wmma_f32_16x16x32_f16.
// Per CDNA5 ISA 7.12.2 (16-bit A 16x32): lane = m + 16*g,
//   halves 0..7  hold K = 8g .. 8g+7
//   halves 8..15 hold K = 16+8g .. 23+8g
// B fragment is the same layout applied to columns (Bt stored [N][K]).
// ---------------------------------------------------------------------------
__device__ __forceinline__ v16h load_frag16(const _Float16* __restrict__ base,
                                            int ld, int row0, int kbase, int lane) {
  const int idx = lane & 15;
  const int g   = lane >> 4;
  const _Float16* p = base + (size_t)(row0 + idx) * ld + kbase + 8 * g;
  v8h lo = *(const v8h*)p;          // K = 8g .. 8g+7
  v8h hi = *(const v8h*)(p + 16);   // K = 16+8g .. 23+8g
  v16h r;
#pragma unroll
  for (int i = 0; i < 8; ++i) { r[i] = lo[i]; r[i + 8] = hi[i]; }
  return r;
}

__device__ __forceinline__ v8f wmma_f16(v16h a, v16h b, v8f c) {
  return __builtin_amdgcn_wmma_f32_16x16x32_f16(false, a, false, b,
                                                (short)0, c, false, false);
}

// ---------------------------------------------------------------------------
// Kernel 1: embedding gather (PAD row forced to zero) + LayerNorm, f16 out.
// One wave (32 lanes) per row of 512 elements; 16 elems/lane via float4.
// ---------------------------------------------------------------------------
__global__ void embed_ln_kernel(const int* __restrict__ src,
                                const float* __restrict__ emb,
                                const float* __restrict__ ln_g,
                                const float* __restrict__ ln_b,
                                _Float16* __restrict__ xnH) {
  const int row  = blockIdx.x * (blockDim.x >> 5) + (threadIdx.x >> 5);
  const int lane = threadIdx.x & 31;
  if (row >= M_ROWS) return;

  const int tok = src[row];
  const bool is_pad = (tok == PAD_IDX);
  const float* erow = emb + (size_t)tok * EE;
  const int e0 = lane * 16;

  float x[16];
#pragma unroll
  for (int j = 0; j < 4; ++j) {
    v4f v = is_pad ? (v4f){0.f, 0.f, 0.f, 0.f}
                   : *(const v4f*)(erow + e0 + j * 4);
#pragma unroll
    for (int k = 0; k < 4; ++k) x[j * 4 + k] = v[k];
  }

  float s = 0.f, sq = 0.f;
#pragma unroll
  for (int i = 0; i < 16; ++i) { s += x[i]; sq += x[i] * x[i]; }
#pragma unroll
  for (int off = 16; off > 0; off >>= 1) {
    s  += __shfl_xor(s, off, 32);
    sq += __shfl_xor(sq, off, 32);
  }
  const float mean = s * (1.f / EE);
  const float var  = sq * (1.f / EE) - mean * mean;
  const float rs   = rsqrtf(var + 1e-5f);

  _Float16* orow = xnH + (size_t)row * EE + e0;
#pragma unroll
  for (int i = 0; i < 16; ++i) {
    const float y = (x[i] - mean) * rs * ln_g[e0 + i] + ln_b[e0 + i];
    orow[i] = (_Float16)y;
  }
}

// ---------------------------------------------------------------------------
// Weight prep: W (K x N row-major, f32) -> Wt (N x K, f16)  [transpose]
//              and plain f32 -> f16 conversion.
// ---------------------------------------------------------------------------
__global__ void transpose_to_f16(const float* __restrict__ W,
                                 _Float16* __restrict__ Wt, int K, int N) {
  const int tid = blockIdx.x * blockDim.x + threadIdx.x;
  if (tid >= K * N) return;
  const int n = tid / K;
  const int k = tid - n * K;
  Wt[tid] = (_Float16)W[(size_t)k * N + n];
}

__global__ void convert_f16(const float* __restrict__ W,
                            _Float16* __restrict__ Wh, int n) {
  const int tid = blockIdx.x * blockDim.x + threadIdx.x;
  if (tid < n) Wh[tid] = (_Float16)W[tid];
}

// ---------------------------------------------------------------------------
// Kernel 2: U = xn(16384x512) * W(512x2048), WMMA f16->f32, f16 output.
// One wave computes a 32x64 tile. A/B working sets (16MB + 2MB f16) are fully
// L2-resident on MI455X (192MB L2) -> direct global fragment loads.
// ---------------------------------------------------------------------------
__global__ void gemm_u_kernel(const _Float16* __restrict__ A,   // M x K (K=512)
                              const _Float16* __restrict__ Bt,  // N x K
                              _Float16* __restrict__ U) {       // M x N f16
  const int KK = EE;            // 512
  const int NN = N_U;           // 2048
  const int ntiles = NN / 64;   // 32
  const int wave = blockIdx.x * (blockDim.x >> 5) + (threadIdx.x >> 5);
  const int lane = threadIdx.x & 31;
  if (wave >= (M_ROWS / 32) * ntiles) return;
  const int row0 = (wave / ntiles) * 32;
  const int col0 = (wave % ntiles) * 64;

  v8f c[2][4];
#pragma unroll
  for (int mi = 0; mi < 2; ++mi)
#pragma unroll
    for (int ni = 0; ni < 4; ++ni) c[mi][ni] = (v8f){0,0,0,0,0,0,0,0};

  for (int kb = 0; kb < KK; kb += 32) {
    v16h a0 = load_frag16(A, KK, row0,      kb, lane);
    v16h a1 = load_frag16(A, KK, row0 + 16, kb, lane);
    v16h b0 = load_frag16(Bt, KK, col0,      kb, lane);
    v16h b1 = load_frag16(Bt, KK, col0 + 16, kb, lane);
    v16h b2 = load_frag16(Bt, KK, col0 + 32, kb, lane);
    v16h b3 = load_frag16(Bt, KK, col0 + 48, kb, lane);
    c[0][0] = wmma_f16(a0, b0, c[0][0]);
    c[0][1] = wmma_f16(a0, b1, c[0][1]);
    c[0][2] = wmma_f16(a0, b2, c[0][2]);
    c[0][3] = wmma_f16(a0, b3, c[0][3]);
    c[1][0] = wmma_f16(a1, b0, c[1][0]);
    c[1][1] = wmma_f16(a1, b1, c[1][1]);
    c[1][2] = wmma_f16(a1, b2, c[1][2]);
    c[1][3] = wmma_f16(a1, b3, c[1][3]);
  }

  // C/D layout: VGPR r, lane l -> (M = r + 8*(l>>4), N = l & 15)
  const int g  = lane >> 4;
  const int nl = lane & 15;
#pragma unroll
  for (int mi = 0; mi < 2; ++mi)
#pragma unroll
    for (int ni = 0; ni < 4; ++ni)
#pragma unroll
      for (int r = 0; r < 8; ++r) {
        const int mrow = row0 + mi * 16 + r + 8 * g;
        const int ncol = col0 + ni * 16 + nl;
        U[(size_t)mrow * NN + ncol] = (_Float16)c[mi][ni][r];
      }
}

// ---------------------------------------------------------------------------
// Kernel 3: SRU scan, one thread per (dir, b, d) recurrence chain.
// U layout per row: [x_tilde | uf | ur | xp] each of width D.
// Writes O (T,B,2D) f16 and final cell state into hidden (B,2D) f32.
// ---------------------------------------------------------------------------
__global__ void sru_scan_kernel(const _Float16* __restrict__ U0,
                                const _Float16* __restrict__ U1,
                                const float* __restrict__ v0,
                                const float* __restrict__ v1,
                                const float* __restrict__ b0,
                                const float* __restrict__ b1,
                                _Float16* __restrict__ O,
                                float* __restrict__ hidden) {
  const int tid = blockIdx.x * blockDim.x + threadIdx.x;
  if (tid >= 2 * BB * DD) return;
  const int dir = tid >> 14;          // / (B*D)
  const int r   = tid & (BB * DD - 1);
  const int b   = r >> 9;             // / D
  const int d   = r & (DD - 1);

  const _Float16* __restrict__ U  = dir ? U1 : U0;
  const float*    __restrict__ vv = dir ? v1 : v0;
  const float*    __restrict__ bbp = dir ? b1 : b0;
  const float vf = vv[d], vr = vv[DD + d];
  const float bf = bbp[d], br = bbp[DD + d];

  float c = 0.f;
  for (int s = 0; s < TT; ++s) {
    const int t = dir ? (TT - 1 - s) : s;
    const size_t base = ((size_t)t * BB + b) * N_U + d;
    const float xt = (float)U[base];
    const float uf = (float)U[base + DD];
    const float ur = (float)U[base + 2 * DD];
    const float xp = (float)U[base + 3 * DD];
    const float f  = 1.f / (1.f + __expf(-(uf + vf * c + bf)));
    const float rg = 1.f / (1.f + __expf(-(ur + vr * c + br)));
    c = f * c + (1.f - f) * xt;
    const float h = rg * c + (1.f - rg) * xp;
    O[((size_t)t * BB + b) * K_OUT + dir * DD + d] = (_Float16)h;
  }
  hidden[(size_t)b * K_OUT + dir * DD + d] = c;
}

// ---------------------------------------------------------------------------
// Kernel 4: out = tanh(O(16384x1024) * W_out^T + b_out), fp32 output.
// W_out is (D, 2D) row-major: row d is K-contiguous -> use directly as Bt.
// ---------------------------------------------------------------------------
__global__ void gemm_out_kernel(const _Float16* __restrict__ O,     // M x 1024
                                const _Float16* __restrict__ WoutH, // 512 x 1024
                                const float* __restrict__ b_out,
                                float* __restrict__ out) {          // M x 512
  const int KK = K_OUT;          // 1024
  const int NN = N_OUT;          // 512
  const int ntiles = NN / 64;    // 8
  const int wave = blockIdx.x * (blockDim.x >> 5) + (threadIdx.x >> 5);
  const int lane = threadIdx.x & 31;
  if (wave >= (M_ROWS / 32) * ntiles) return;
  const int row0 = (wave / ntiles) * 32;
  const int col0 = (wave % ntiles) * 64;

  v8f c[2][4];
#pragma unroll
  for (int mi = 0; mi < 2; ++mi)
#pragma unroll
    for (int ni = 0; ni < 4; ++ni) c[mi][ni] = (v8f){0,0,0,0,0,0,0,0};

  for (int kb = 0; kb < KK; kb += 32) {
    v16h a0 = load_frag16(O, KK, row0,      kb, lane);
    v16h a1 = load_frag16(O, KK, row0 + 16, kb, lane);
    v16h b0 = load_frag16(WoutH, KK, col0,      kb, lane);
    v16h b1 = load_frag16(WoutH, KK, col0 + 16, kb, lane);
    v16h b2 = load_frag16(WoutH, KK, col0 + 32, kb, lane);
    v16h b3 = load_frag16(WoutH, KK, col0 + 48, kb, lane);
    c[0][0] = wmma_f16(a0, b0, c[0][0]);
    c[0][1] = wmma_f16(a0, b1, c[0][1]);
    c[0][2] = wmma_f16(a0, b2, c[0][2]);
    c[0][3] = wmma_f16(a0, b3, c[0][3]);
    c[1][0] = wmma_f16(a1, b0, c[1][0]);
    c[1][1] = wmma_f16(a1, b1, c[1][1]);
    c[1][2] = wmma_f16(a1, b2, c[1][2]);
    c[1][3] = wmma_f16(a1, b3, c[1][3]);
  }

  const int g  = lane >> 4;
  const int nl = lane & 15;
#pragma unroll
  for (int mi = 0; mi < 2; ++mi)
#pragma unroll
    for (int ni = 0; ni < 4; ++ni)
#pragma unroll
      for (int r = 0; r < 8; ++r) {
        const int mrow = row0 + mi * 16 + r + 8 * g;
        const int ncol = col0 + ni * 16 + nl;
        out[(size_t)mrow * NN + ncol] = tanhf(c[mi][ni][r] + b_out[ncol]);
      }
}

// ---------------------------------------------------------------------------
// Host-side launch sequence (graph-capture safe: no alloc/sync).
// ---------------------------------------------------------------------------
extern "C" void kernel_launch(void* const* d_in, const int* in_sizes, int n_in,
                              void* d_out, int out_size, void* d_ws, size_t ws_size,
                              hipStream_t stream) {
  const int*   src   = (const int*)  d_in[0];
  const float* emb   = (const float*)d_in[1];
  const float* ln_g  = (const float*)d_in[2];
  const float* ln_b  = (const float*)d_in[3];
  const float* W0    = (const float*)d_in[4];
  const float* W1    = (const float*)d_in[5];
  const float* v0    = (const float*)d_in[6];
  const float* v1    = (const float*)d_in[7];
  const float* b0    = (const float*)d_in[8];
  const float* b1    = (const float*)d_in[9];
  const float* W_out = (const float*)d_in[10];
  const float* b_out = (const float*)d_in[11];

  float* out    = (float*)d_out;                        // (T,B,D)
  float* hidden = (float*)d_out + (size_t)M_ROWS * DD;  // (1,B,2D)

  // Workspace carve-up (bytes)
  char* ws = (char*)d_ws;
  _Float16* xnH   = (_Float16*)(ws);                                   // 16 MB
  _Float16* W0t   = (_Float16*)(ws + (size_t)16 * 1024 * 1024);        //  2 MB
  _Float16* W1t   = (_Float16*)(ws + (size_t)18 * 1024 * 1024);        //  2 MB
  _Float16* WoutH = (_Float16*)(ws + (size_t)20 * 1024 * 1024);        //  1 MB
  _Float16* U0    = (_Float16*)(ws + (size_t)22 * 1024 * 1024);        // 64 MB
  _Float16* U1    = (_Float16*)(ws + (size_t)86 * 1024 * 1024);        // 64 MB
  _Float16* Obuf  = (_Float16*)(ws + (size_t)150 * 1024 * 1024);       // 32 MB

  // 1) embedding + LayerNorm -> xn (f16).  One wave per row, 8 waves/block.
  embed_ln_kernel<<<M_ROWS / 8, 256, 0, stream>>>(src, emb, ln_g, ln_b, xnH);

  // 2) weight prep
  transpose_to_f16<<<(EE * N_U + 255) / 256, 256, 0, stream>>>(W0, W0t, EE, N_U);
  transpose_to_f16<<<(EE * N_U + 255) / 256, 256, 0, stream>>>(W1, W1t, EE, N_U);
  convert_f16<<<(DD * K_OUT + 255) / 256, 256, 0, stream>>>(W_out, WoutH, DD * K_OUT);

  // 3) U GEMMs: (16384/32)*(2048/64) = 16384 waves; 8 waves per 256-thread block
  const int waves_u = (M_ROWS / 32) * (N_U / 64);
  gemm_u_kernel<<<waves_u / 8, 256, 0, stream>>>(xnH, W0t, U0);
  gemm_u_kernel<<<waves_u / 8, 256, 0, stream>>>(xnH, W1t, U1);

  // 4) SRU scan: 2*B*D = 32768 threads
  sru_scan_kernel<<<(2 * BB * DD) / 256, 256, 0, stream>>>(U0, U1, v0, v1, b0, b1,
                                                           Obuf, hidden);

  // 5) output GEMM + bias + tanh: (16384/32)*(512/64) = 4096 waves
  const int waves_o = (M_ROWS / 32) * (N_OUT / 64);
  gemm_out_kernel<<<waves_o / 8, 256, 0, stream>>>(Obuf, WoutH, b_out, out);
}
